// Net_81638738362653
// MI455X (gfx1250) — matrix operand order repaired and checked by reading, not compiled, not run on hardware
//
#include <hip/hip_runtime.h>
#include <hip/hip_bf16.h>

typedef __attribute__((ext_vector_type(16))) _Float16 v16h;
typedef __attribute__((ext_vector_type(8)))  _Float16 v8h;
typedef __attribute__((ext_vector_type(8)))  float    v8f;

#define T_STEPS 784
#define BATCH   512
#define HIDDEN  128
#define NCLS    10
#define M_TOTAL (T_STEPS * BATCH)   // 401408 rows for the big GEMM

// ---------------------------------------------------------------------------
// Kernel 0: convert W2 (f32 [128][128], row-major, W2[ho][hi]) to f16
// ---------------------------------------------------------------------------
__global__ void k_cvt_w2(const float* __restrict__ w2, _Float16* __restrict__ w2h) {
    int i = blockIdx.x * 256 + threadIdx.x;
    if (i < HIDDEN * HIDDEN) w2h[i] = (_Float16)w2[i];
}

// ---------------------------------------------------------------------------
// Kernel 1: layer-1 IndRNN recurrence. One thread per (b,h); sequential over t.
// Writes h1 as f16, layout [t*BATCH + b][h] (row-major, h contiguous) so it is
// directly the A matrix of the layer-2 GEMM.
// ---------------------------------------------------------------------------
__global__ void k_layer1(const float* __restrict__ x,   // [T][B][1]
                         const float* __restrict__ W1,  // [H][1]
                         const float* __restrict__ b1,  // [H]
                         const float* __restrict__ u1,  // [H]
                         _Float16* __restrict__ h1) {   // [T*B][H] f16
    int tid = blockIdx.x * 256 + threadIdx.x;           // 65536 threads
    int h = tid & (HIDDEN - 1);
    int b = tid >> 7;
    float w  = W1[h];
    float bb = b1[h];
    float uu = u1[h];
    float hs = 0.0f;
    for (int t = 0; t < T_STEPS; ++t) {
        float pre = x[t * BATCH + b] * w + bb;
        hs = pre + uu * hs;
        hs = hs > 0.0f ? hs : 0.0f;
        h1[(size_t)(t * BATCH + b) * HIDDEN + h] = (_Float16)hs;
    }
}

// ---------------------------------------------------------------------------
// Kernel 2: pre2 = h1 @ W2^T  via v_wmma_f32_16x16x32_f16.
// C[m][n] = sum_k A[m][k] * W2[n][k]  (so WMMA-B[k][n] = W2[n][k]; each B lane
// reads a contiguous 16-f16 segment of a W2 row).
// One wave computes a 16(M) x 128(N) strip, K=128 in 4 chunks of 32.
// All 8 B tiles of a k-chunk are staged into registers first so the compiler
// can issue one load clause and stagger s_wait_loadcnt across the 8 WMMAs.
// ---------------------------------------------------------------------------
__global__ void __launch_bounds__(256)
k_gemm_pre2(const _Float16* __restrict__ A,   // h1 f16 [M_TOTAL][128]
            const _Float16* __restrict__ Bw,  // W2 f16 [128][128]
            float* __restrict__ C) {          // pre2 f32 [M_TOTAL][128]
    const int wave  = threadIdx.x >> 5;
    const int lane  = threadIdx.x & 31;
    const int mBase = (blockIdx.x * 8 + wave) * 16;
    const int lo    = lane & 15;
    const int grpA  = (lane >> 4) ? 16 : 0;   // A: K sub-offset per lane group
    const int grpB  = (lane >> 4) ? 16 : 0;   // B: K sub-offset per lane group

    v8f acc[8] = {};  // 8 N-tiles of 16x16 f32

    const _Float16* aRow = A + (size_t)(mBase + lo) * HIDDEN + grpA;
    const _Float16* bRow = Bw + (size_t)lo * HIDDEN + grpB;

#pragma unroll
    for (int kc = 0; kc < 4; ++kc) {
        const int k0 = kc * 32;
        // A 16x32 f16 tile: lane holds K = k0+grpA+[0..15]
        v8h a0 = *(const v8h*)(aRow + k0);
        v8h a1 = *(const v8h*)(aRow + k0 + 8);
        v16h a = __builtin_shufflevector(a0, a1,
                                         0, 1, 2, 3, 4, 5, 6, 7,
                                         8, 9, 10, 11, 12, 13, 14, 15);
        // Stage all 8 B tiles (one clause of 16 b128 loads), then 8 WMMAs.
        v16h bv[8];
#pragma unroll
        for (int nt = 0; nt < 8; ++nt) {
            bv[nt] = *(const v16h*)(bRow + (size_t)(nt * 16) * HIDDEN + k0);
        }
#pragma unroll
        for (int nt = 0; nt < 8; ++nt) {
            acc[nt] = __builtin_amdgcn_wmma_f32_16x16x32_f16(
                false, a, false, bv[nt], (short)0, acc[nt], false, false);
        }
    }

    // Store: C/D layout -> VGPR r: M = r + 8*(lane>=16), N = lane&15
    const int rHi = (lane >> 4) * 8;
#pragma unroll
    for (int nt = 0; nt < 8; ++nt) {
#pragma unroll
        for (int r = 0; r < 8; ++r) {
            C[(size_t)(mBase + rHi + r) * HIDDEN + nt * 16 + lo] = acc[nt][r];
        }
    }
}

// ---------------------------------------------------------------------------
// Kernel 3: layer-2 IndRNN recurrence over pre2 (streaming, coalesced).
// One thread per (b,h); keeps only the final hidden state.
// ---------------------------------------------------------------------------
__global__ void k_layer2(const float* __restrict__ pre2, // [T*B][H]
                         const float* __restrict__ b2,   // [H]
                         const float* __restrict__ u2,   // [H]
                         float* __restrict__ h2last) {   // [B][H]
    int tid = blockIdx.x * 256 + threadIdx.x;            // 65536 threads
    int h = tid & (HIDDEN - 1);
    int b = tid >> 7;
    float bb = b2[h];
    float uu = u2[h];
    float hs = 0.0f;
    for (int t = 0; t < T_STEPS; ++t) {
        float pre = pre2[(size_t)(t * BATCH + b) * HIDDEN + h] + bb;
        hs = pre + uu * hs;
        hs = hs > 0.0f ? hs : 0.0f;
    }
    h2last[tid] = hs;
}

// ---------------------------------------------------------------------------
// Kernel 4: classifier head, out[b][c] = h2last[b]·lin_W[c] + lin_b[c]
// ---------------------------------------------------------------------------
__global__ void k_head(const float* __restrict__ h2,     // [B][H]
                       const float* __restrict__ lin_W,  // [10][H]
                       const float* __restrict__ lin_b,  // [10]
                       float* __restrict__ out) {        // [B][10]
    int tid = blockIdx.x * 256 + threadIdx.x;
    if (tid >= BATCH * NCLS) return;
    int c = tid % NCLS;
    int b = tid / NCLS;
    float s = lin_b[c];
    const float* hp = h2 + b * HIDDEN;
    const float* wp = lin_W + c * HIDDEN;
#pragma unroll 8
    for (int h = 0; h < HIDDEN; ++h) s += hp[h] * wp[h];
    out[tid] = s;
}

// ---------------------------------------------------------------------------
extern "C" void kernel_launch(void* const* d_in, const int* in_sizes, int n_in,
                              void* d_out, int out_size, void* d_ws, size_t ws_size,
                              hipStream_t stream) {
    const float* x     = (const float*)d_in[0];
    const float* W1    = (const float*)d_in[1];
    const float* b1    = (const float*)d_in[2];
    const float* u1    = (const float*)d_in[3];
    const float* W2    = (const float*)d_in[4];
    const float* b2    = (const float*)d_in[5];
    const float* u2    = (const float*)d_in[6];
    const float* lin_W = (const float*)d_in[7];
    const float* lin_b = (const float*)d_in[8];
    float* out = (float*)d_out;

    char* ws = (char*)d_ws;
    // Workspace layout (all 256B-aligned):
    const size_t h1_bytes   = (size_t)M_TOTAL * HIDDEN * sizeof(_Float16); // 102,760,448
    const size_t pre2_bytes = (size_t)M_TOTAL * HIDDEN * sizeof(float);    // 205,520,896
    _Float16* h1     = (_Float16*)(ws);
    float*    pre2   = (float*)(ws + h1_bytes);
    _Float16* w2h    = (_Float16*)(ws + h1_bytes + pre2_bytes);
    float*    h2last = (float*)(ws + h1_bytes + pre2_bytes + 32768);

    // 0) W2 -> f16 (16384 elems)
    k_cvt_w2<<<64, 256, 0, stream>>>(W2, w2h);
    // 1) layer-1 recurrence: 65536 threads
    k_layer1<<<256, 256, 0, stream>>>(x, W1, b1, u1, h1);
    // 2) GEMM: M_TOTAL/16 = 25088 waves, 8 waves/block -> 3136 blocks
    k_gemm_pre2<<<M_TOTAL / 16 / 8, 256, 0, stream>>>(h1, w2h, pre2);
    // 3) layer-2 recurrence: 65536 threads
    k_layer2<<<256, 256, 0, stream>>>(pre2, b2, u2, h2last);
    // 4) head: 5120 outputs
    k_head<<<(BATCH * NCLS + 255) / 256, 256, 0, stream>>>(h2last, lin_W, lin_b, out);
}
